// MLASelfAttentionBlock_24893630447987
// MI455X (gfx1250) — compile-verified, matrix operand order checked by reading, not run on hardware
//
#include <hip/hip_runtime.h>
#include <hip/hip_bf16.h>
#include <math.h>

// ---------------- MLA dimensions (compile-time) ----------------
#define D_MODEL   2048
#define D_LATENT  512
#define NUM_HEADS 16
#define HEAD_DIM  128            // D_MODEL / NUM_HEADS
#define BATCHSZ   2
#define SEQ       2048
#define BT        (BATCHSZ * SEQ)

typedef __attribute__((ext_vector_type(16))) _Float16     v16h;
typedef __attribute__((ext_vector_type(8)))  _Float16     v8h;
typedef __attribute__((ext_vector_type(8)))  float        v8f;
typedef __attribute__((ext_vector_type(4)))  unsigned int v4u;

// ---------------------------------------------------------------
// CDNA5 async global->LDS copy (ASYNCcnt-tracked, bypasses VGPRs).
// GV mode: per-lane 64-bit global address; VDST VGPR carries the LDS
// byte address (low 32 bits of the generic pointer to __shared__).
// Both addresses must be 16B aligned for b128.
// ---------------------------------------------------------------
__device__ __forceinline__ void async_b128(_Float16* lds_dst, const _Float16* gsrc) {
  const unsigned lds_off = (unsigned)(unsigned long long)(void*)lds_dst;
  asm volatile("global_load_async_to_lds_b128 %0, %1, off"
               :: "v"(lds_off), "v"(gsrc)
               : "memory");
}
__device__ __forceinline__ void wait_async0() {
  asm volatile("s_wait_asynccnt 0" ::: "memory");
}

// ---------------------------------------------------------------
// Fragment loader for CDNA5 16-bit WMMA operands (wave32).
// A-matrix 16x32 layout (ISA 7.12.2): lane L holds row M=L&15;
// halves j map K = j + (j>=8?8:0) + (L>=16?8:0) -> the 8 dwords are two
// contiguous 16B runs (uint idx 0..3 and 8..11): two ds_load_b128.
// B tiles are staged N-major in LDS so the same loader serves B.
// Row strides used (40, 520 halves) are multiples of 8 -> 16B aligned.
// ---------------------------------------------------------------
__device__ __forceinline__ v16h load_frag(const _Float16* __restrict__ base,
                                          int row_stride_halves, int lane) {
  const _Float16* p = base + (size_t)(lane & 15) * row_stride_halves
                           + ((lane >> 4) << 3);           // +8 halves for lanes 16..31
  const v4u* pv = (const v4u*)p;
  union { v16h v; v4u q[2]; } r;
  r.q[0] = pv[0];   // K pattern low run  (bytes 0..15)
  r.q[1] = pv[2];   // K pattern high run (bytes 32..47)
  return r.v;
}

__device__ __forceinline__ v8f wmma16(v16h a, v16h b, v8f c) {
  return __builtin_amdgcn_wmma_f32_16x16x32_f16(false, a, false, b,
                                                (short)0, c, false, false);
}

// ---------------- f32 -> f16 cast ----------------
__global__ void cast_f32_f16(const float* __restrict__ s, _Float16* __restrict__ d, int n) {
  int i = blockIdx.x * blockDim.x + threadIdx.x;
  if (i < n) d[i] = (_Float16)s[i];
}

// ---------------------------------------------------------------
// Generic batched WMMA GEMM: C[bz] = A[bz](MxK,row-major) @ B[bz](KxN,row-major)
// f16 in, f32 accumulate. Block tile 128x128, BK=32, 256 threads = 8 wave32s,
// each wave a 32x64 sub-tile (2x4 WMMA C-tiles).
// Double-buffered LDS; A staged via async-to-LDS DMA, B via b128 global reads
// + u16 transpose scatter (reads issued before the WMMAs, stores sunk after).
// Requires M%128==0, N%128==0, K%32==0 (true for all calls here).
// ---------------------------------------------------------------
template <bool OUT_F32>
__global__ __launch_bounds__(256)
void gemm_f16(const _Float16* __restrict__ A, const _Float16* __restrict__ B,
              void* __restrict__ Cout,
              int M, int N, int K, int lda, int ldb, int ldc,
              long long strideA, long long strideB, long long strideC) {
  __shared__ _Float16 As[2][128 * 40];   // [m][k], stride 40 halves (80B = 5x16B)
  __shared__ _Float16 Bs[2][128 * 40];   // [n][k] (N-major)

  const int bz = blockIdx.z;
  const _Float16* Ab = A + (size_t)bz * strideA;
  const _Float16* Bb = B + (size_t)bz * strideB;

  const int row0 = blockIdx.y * 128;
  const int col0 = blockIdx.x * 128;
  const int tid  = threadIdx.x;
  const int lane = tid & 31;
  const int wid  = tid >> 5;
  const int wm   = wid & 3;   // 4 waves along M -> 32 rows each
  const int wn   = wid >> 2;  // 2 waves along N -> 64 cols each

  v8h bchunk[2];

  auto issueA = [&](int k0, int buf) {       // 512 x 16B chunks, 2 per thread
#pragma unroll
    for (int j = 0; j < 2; ++j) {
      const int c = tid + j * 256;
      const int m = c >> 2, c16 = c & 3;
      async_b128(&As[buf][m * 40 + c16 * 8],
                 Ab + (size_t)(row0 + m) * lda + k0 + c16 * 8);
    }
  };
  auto loadB = [&](int k0) {                 // 2 x global_load_b128 per thread
#pragma unroll
    for (int j = 0; j < 2; ++j) {
      const int c = tid + j * 256;           // 32 rows x 16 chunks
      const int k = c >> 4, nc = c & 15;
      bchunk[j] = *(const v8h*)(Bb + (size_t)(k0 + k) * ldb + col0 + nc * 8);
    }
  };
  auto storeB = [&](int buf) {               // transpose scatter into N-major LDS
#pragma unroll
    for (int j = 0; j < 2; ++j) {
      const int c = tid + j * 256;
      const int k = c >> 4, nc = c & 15;
#pragma unroll
      for (int e = 0; e < 8; ++e)
        Bs[buf][(nc * 8 + e) * 40 + k] = bchunk[j][e];
    }
  };

  v8f acc[2][4];
#pragma unroll
  for (int i = 0; i < 2; ++i)
#pragma unroll
    for (int j = 0; j < 4; ++j) acc[i][j] = (v8f){};

  // pipeline prologue
  issueA(0, 0);
  loadB(0);
  storeB(0);
  wait_async0();
  __syncthreads();

  const int nstages = K / 32;
  for (int s = 0; s < nstages; ++s) {
    const int cur = s & 1, nxt = cur ^ 1;
    const bool has_next = (s + 1 < nstages);
    if (has_next) {                 // overlap next-stage fetch with WMMAs
      issueA((s + 1) * 32, nxt);
      loadB((s + 1) * 32);
    }

    const v16h a0 = load_frag(&As[cur][(wm * 32 + 0)  * 40], 40, lane);
    const v16h a1 = load_frag(&As[cur][(wm * 32 + 16) * 40], 40, lane);
#pragma unroll
    for (int j = 0; j < 4; ++j) {
      const v16h b = load_frag(&Bs[cur][(wn * 64 + j * 16) * 40], 40, lane);
      acc[0][j] = wmma16(a0, b, acc[0][j]);
      acc[1][j] = wmma16(a1, b, acc[1][j]);
    }

    if (has_next) storeB(nxt);      // LOADcnt wait lands here, behind compute
    wait_async0();
    __syncthreads();
  }

  // C-tile layout: VGPR r, lane l -> M = r + (l>=16 ? 8 : 0), N = l&15.
  const int hi8 = (lane >> 4) * 8;
  const int nl  = lane & 15;
#pragma unroll
  for (int mi = 0; mi < 2; ++mi)
#pragma unroll
    for (int j = 0; j < 4; ++j)
#pragma unroll
      for (int r = 0; r < 8; ++r) {
        const int row = row0 + wm * 32 + mi * 16 + r + hi8;
        const int col = col0 + wn * 64 + j * 16 + nl;
        if (OUT_F32) {
          ((float*)Cout)[(size_t)bz * strideC + (size_t)row * ldc + col] = acc[mi][j][r];
        } else {
          ((_Float16*)Cout)[(size_t)bz * strideC + (size_t)row * ldc + col] =
              (_Float16)acc[mi][j][r];
        }
      }
}

// ---------------------------------------------------------------
// Flash attention over latent space, causal.
//   scores[t,s] = (q_lat[b,h,t,:] . kv_latent[b,s,:]) * scale   (L = 512)
//   y[b,t,h,:]  = softmax_s(scores) @ v[b,s,h,:]                (S = 128)
// Grid: (T/128, H, B); 256 threads = 8 waves; each wave owns 16 query rows.
// LDS ~225 KB (CDNA5 320KB WGP). All LDS addressing is smem + integer
// offset (buffer select on *integers*, not pointers) so address-space
// inference keeps every access on the DS path (no flat_load on LDS).
// ---------------------------------------------------------------
#define QSTR 520   // 512 + 8 pad halves (1040B = 65x16B)
#define VSTR 40    // 32 + 8 pad
#define QS_OFF 0
#define KS_OFF (128 * QSTR)                       // + cur * (32*QSTR)
#define VT_OFF (KS_OFF + 2 * 32 * QSTR)           // + cur * (128*VSTR)
#define PS_OFF (VT_OFF + 2 * 128 * VSTR)          // 8 waves x 16 x VSTR
#define SMEM_HALVES (PS_OFF + 8 * 16 * VSTR)

__global__ __launch_bounds__(256)
void mla_flash_attn(const _Float16* __restrict__ qlat,   // [H][B*T][L]
                    const _Float16* __restrict__ kvlat,  // [B*T][L]
                    const _Float16* __restrict__ vh,     // [B*T][D_MODEL]
                    _Float16* __restrict__ yh,           // [B*T][D_MODEL]
                    float scale) {
  extern __shared__ _Float16 smem[];

  const int q0 = blockIdx.x * 128;
  const int h  = blockIdx.y;
  const int b  = blockIdx.z;
  const int tid  = threadIdx.x;
  const int lane = tid & 31;
  const int wv   = tid >> 5;

  v8h vchunk[2];

  const size_t qbase = (size_t)h * BT * D_LATENT + (size_t)(b * SEQ + q0) * D_LATENT;
  auto issueQ = [&]() {                        // 8192 x 16B chunks, 32 per thread
#pragma unroll
    for (int j = 0; j < 32; ++j) {
      const int c = tid + j * 256;
      const int m = c >> 6, c16 = c & 63;
      async_b128(smem + QS_OFF + m * QSTR + c16 * 8,
                 qlat + qbase + (size_t)m * D_LATENT + c16 * 8);
    }
  };
  auto issueK = [&](int s0, int buf) {         // 2048 x 16B chunks, 8 per thread
#pragma unroll
    for (int j = 0; j < 8; ++j) {
      const int c = tid + j * 256;
      const int s = c >> 6, c16 = c & 63;
      async_b128(smem + KS_OFF + buf * (32 * QSTR) + s * QSTR + c16 * 8,
                 kvlat + (size_t)(b * SEQ + s0 + s) * D_LATENT + c16 * 8);
    }
  };
  auto loadV = [&](int s0) {                   // 2 x global_load_b128 per thread
#pragma unroll
    for (int j = 0; j < 2; ++j) {
      const int c = tid + j * 256;             // 32 rows x 16 chunks
      const int s = c >> 4, dc = c & 15;
      vchunk[j] = *(const v8h*)(vh + (size_t)(b * SEQ + s0 + s) * D_MODEL
                                + h * HEAD_DIM + dc * 8);
    }
  };
  auto storeV = [&](int buf) {                 // transpose scatter into [d][s]
#pragma unroll
    for (int j = 0; j < 2; ++j) {
      const int c = tid + j * 256;
      const int s = c >> 4, dc = c & 15;
#pragma unroll
      for (int e = 0; e < 8; ++e)
        smem[VT_OFF + buf * (128 * VSTR) + (dc * 8 + e) * VSTR + s] = vchunk[j][e];
    }
  };

  v8f acc[8];
#pragma unroll
  for (int d0 = 0; d0 < 8; ++d0) acc[d0] = (v8f){};
  float m_r[8], l_r[8];
#pragma unroll
  for (int r = 0; r < 8; ++r) { m_r[r] = -3.0e38f; l_r[r] = 0.0f; }

  const int t_base = q0 + wv * 16;
  const int hi8 = (lane >> 4) * 8;
  const int nl  = lane & 15;
  const int nblk = q0 / 32 + 4;   // causal: key blocks up to and incl. diagonal

  // pipeline prologue: Q + first K/V block
  issueQ();
  issueK(0, 0);
  loadV(0);
  storeV(0);
  wait_async0();
  __syncthreads();

  for (int i = 0; i < nblk; ++i) {
    const int s0 = i * 32;
    const int cur = i & 1, nxt = cur ^ 1;
    const bool has_next = (i + 1 < nblk);
    if (has_next) {                 // overlap next block fetch with 40 WMMAs
      issueK((i + 1) * 32, nxt);
      loadV((i + 1) * 32);
    }
    const int koff = KS_OFF + cur * (32 * QSTR);
    const int voff = VT_OFF + cur * (128 * VSTR);

    // Scores: 16x32 tile pair over L=512 -> 32 WMMAs
    v8f S0 = (v8f){}, S1 = (v8f){};
#pragma unroll
    for (int kc = 0; kc < D_LATENT / 32; ++kc) {
      const v16h aq  = load_frag(smem + QS_OFF + (wv * 16) * QSTR + kc * 32, QSTR, lane);
      const v16h bk0 = load_frag(smem + koff + 0  * QSTR + kc * 32, QSTR, lane);
      const v16h bk1 = load_frag(smem + koff + 16 * QSTR + kc * 32, QSTR, lane);
      S0 = wmma16(aq, bk0, S0);
      S1 = wmma16(aq, bk1, S1);
    }

    // scale + causal mask + online softmax.  Each C-VGPR r holds row r
    // (lanes 0..15) and row r+8 (lanes 16..31); reduce with xor masks < 16.
#pragma unroll
    for (int r = 0; r < 8; ++r) {
      const int t = t_base + r + hi8;
      float a0 = (float)S0[r] * scale;
      float a1 = (float)S1[r] * scale;
      if (s0 + nl      > t) a0 = -3.0e38f;
      if (s0 + 16 + nl > t) a1 = -3.0e38f;
      float mx = fmaxf(a0, a1);
#pragma unroll
      for (int off = 8; off >= 1; off >>= 1) mx = fmaxf(mx, __shfl_xor(mx, off, 32));
      const float mnew = fmaxf(m_r[r], mx);
      const float p0 = __expf(a0 - mnew);
      const float p1 = __expf(a1 - mnew);
      const float cor = __expf(m_r[r] - mnew);
      float sum = p0 + p1;
#pragma unroll
      for (int off = 8; off >= 1; off >>= 1) sum += __shfl_xor(sum, off, 32);
      l_r[r] = l_r[r] * cor + sum;
      m_r[r] = mnew;
#pragma unroll
      for (int d0 = 0; d0 < 8; ++d0) acc[d0][r] *= cor;
      // P -> per-wave LDS scratch (re-stripe C-layout into A-fragment layout)
      const int pp = PS_OFF + wv * (16 * VSTR) + (r + hi8) * VSTR;
      smem[pp + nl]      = (_Float16)p0;
      smem[pp + 16 + nl] = (_Float16)p1;
    }

    // Wave-private LDS RAW across lanes: drain DS counter before re-reading.
    asm volatile("s_wait_dscnt 0" ::: "memory");

    // PV: P (16x32) x V (32x128) -> 8 WMMAs
    const v16h ap = load_frag(smem + PS_OFF + wv * (16 * VSTR), VSTR, lane);
#pragma unroll
    for (int d0 = 0; d0 < 8; ++d0) {
      const v16h bv = load_frag(smem + voff + (d0 * 16) * VSTR, VSTR, lane);
      acc[d0] = wmma16(ap, bv, acc[d0]);
    }

    if (has_next) storeV(nxt);      // LOADcnt wait lands here, behind compute
    wait_async0();
    __syncthreads();
  }

  // Epilogue: divide by row sum, store y as f16 (consumed by final GEMM)
#pragma unroll
  for (int r = 0; r < 8; ++r) {
    const float inv = 1.0f / l_r[r];
    const int t = t_base + r + hi8;
#pragma unroll
    for (int d0 = 0; d0 < 8; ++d0) {
      yh[(size_t)(b * SEQ + t) * D_MODEL + h * HEAD_DIM + d0 * 16 + nl] =
          (_Float16)(acc[d0][r] * inv);
    }
  }
}

// ---------------------------------------------------------------
extern "C" void kernel_launch(void* const* d_in, const int* in_sizes, int n_in,
                              void* d_out, int out_size, void* d_ws, size_t ws_size,
                              hipStream_t stream) {
  (void)in_sizes; (void)n_in; (void)out_size; (void)ws_size;
  const float* x     = (const float*)d_in[0];
  const float* w_dkv = (const float*)d_in[1];
  const float* w_uk  = (const float*)d_in[2];
  const float* w_uv  = (const float*)d_in[3];
  const float* w_q   = (const float*)d_in[4];
  const float* w_o   = (const float*)d_in[5];
  float* out = (float*)d_out;

  char* ws = (char*)d_ws;
  size_t off = 0;
  auto alloc = [&](size_t halves) {
    _Float16* p = (_Float16*)(ws + off);
    off = (off + halves * sizeof(_Float16) + 255) & ~(size_t)255;
    return p;
  };
  _Float16* xh       = alloc((size_t)BT * D_MODEL);
  _Float16* wdkvh    = alloc((size_t)D_MODEL * D_LATENT);
  _Float16* wukh     = alloc((size_t)D_MODEL * D_LATENT);
  _Float16* wuvh     = alloc((size_t)D_LATENT * D_MODEL);
  _Float16* wqh      = alloc((size_t)D_MODEL * D_MODEL);
  _Float16* woh      = alloc((size_t)D_MODEL * D_MODEL);
  _Float16* kvlat    = alloc((size_t)BT * D_LATENT);
  _Float16* absorbed = alloc((size_t)NUM_HEADS * D_MODEL * D_LATENT);
  _Float16* qlat     = alloc((size_t)NUM_HEADS * BT * D_LATENT);
  _Float16* vh       = alloc((size_t)BT * D_MODEL);
  _Float16* yh       = alloc((size_t)BT * D_MODEL);

  auto cast = [&](const float* s, _Float16* d, size_t n) {
    cast_f32_f16<<<dim3((unsigned)((n + 255) / 256)), dim3(256), 0, stream>>>(s, d, (int)n);
  };
  cast(x,     xh,    (size_t)BT * D_MODEL);
  cast(w_dkv, wdkvh, (size_t)D_MODEL * D_LATENT);
  cast(w_uk,  wukh,  (size_t)D_MODEL * D_LATENT);
  cast(w_uv,  wuvh,  (size_t)D_LATENT * D_MODEL);
  cast(w_q,   wqh,   (size_t)D_MODEL * D_MODEL);
  cast(w_o,   woh,   (size_t)D_MODEL * D_MODEL);

  // 1) kv_latent = x @ w_dkv            [4096,2048]x[2048,512]
  gemm_f16<false><<<dim3(D_LATENT / 128, BT / 128, 1), 256, 0, stream>>>(
      xh, wdkvh, kvlat, BT, D_LATENT, D_MODEL, D_MODEL, D_LATENT, D_LATENT, 0, 0, 0);

  // 2) absorbed[h] = w_q[:, h*128:(h+1)*128] @ w_uk[h*128:(h+1)*128, :]
  gemm_f16<false><<<dim3(D_LATENT / 128, D_MODEL / 128, NUM_HEADS), 256, 0, stream>>>(
      wqh, wukh, absorbed, D_MODEL, D_LATENT, HEAD_DIM, D_MODEL, D_LATENT, D_LATENT,
      (long long)HEAD_DIM, (long long)HEAD_DIM * D_LATENT, (long long)D_MODEL * D_LATENT);

  // 3) q_lat[h] = x @ absorbed[h]        batched over heads, shared A
  gemm_f16<false><<<dim3(D_LATENT / 128, BT / 128, NUM_HEADS), 256, 0, stream>>>(
      xh, absorbed, qlat, BT, D_LATENT, D_MODEL, D_MODEL, D_LATENT, D_LATENT,
      0, (long long)D_MODEL * D_LATENT, (long long)BT * D_LATENT);

  // 4) v = kv_latent @ w_uv              [4096,512]x[512,2048]
  gemm_f16<false><<<dim3(D_MODEL / 128, BT / 128, 1), 256, 0, stream>>>(
      kvlat, wuvh, vh, BT, D_MODEL, D_LATENT, D_LATENT, D_MODEL, D_MODEL, 0, 0, 0);

  // 5) causal flash attention in latent space -> y (f16)
  const float scale = 1.0f / sqrtf((float)D_MODEL);
  const size_t smem_bytes = (size_t)SMEM_HALVES * sizeof(_Float16);
  mla_flash_attn<<<dim3(SEQ / 128, NUM_HEADS, BATCHSZ), 256, smem_bytes, stream>>>(
      qlat, kvlat, vh, yh, scale);

  // 6) out = y @ w_o (f32 straight to d_out)
  gemm_f16<true><<<dim3(D_MODEL / 128, BT / 128, 1), 256, 0, stream>>>(
      yh, woh, out, BT, D_MODEL, D_MODEL, D_MODEL, D_MODEL, D_MODEL, 0, 0, 0);
}
